// GRUVAE_35072702939737
// MI455X (gfx1250) — matrix operand order repaired and checked by reading, not compiled, not run on hardware
//
#include <hip/hip_runtime.h>
#include <hip/hip_bf16.h>

// ---------------------------------------------------------------------------
// Types for CDNA5 WMMA (wave32): v_wmma_f32_16x16x32_bf16
// ---------------------------------------------------------------------------
typedef __attribute__((ext_vector_type(16))) __bf16 v16bf;
typedef __attribute__((ext_vector_type(4)))  __bf16 v4bf;
typedef __attribute__((ext_vector_type(8)))  float  v8f;

#define T_ALL   4096
#define T_M1    4095

// ---------------------------------------------------------------------------
// Generic WMMA GEMM:  C[M,N] = act( A[M,K] * B + bias )
//   A: f32 row-major, lda = K (multiple of 64), converted f32->bf16 into LDS.
//   B: f32; transB==1 -> B is (N x KB) row-major (used only when K == KB)
//           transB==0 -> B is (KB x N) row-major (A zero-padded for k >= KB)
//   Block: 256 threads = 8 waves; block tile 64(M) x 64(N); K step 64.
//   Wave grid 4(M) x 2(N); each wave: two 16x16 f32 accumulators,
//   4 v_wmma per K-round. Double-buffered LDS: global loads for tile k+64
//   are issued before the WMMAs on tile k, converted/stored after -> load
//   latency hidden behind matrix math, one barrier per round.
// ---------------------------------------------------------------------------
__global__ __launch_bounds__(256)
void gvae_wmma_gemm(const float* __restrict__ A, const float* __restrict__ B,
                    const float* __restrict__ bias, float* __restrict__ C,
                    int M, int N, int K, int KB, int transB, int relu)
{
    __shared__ __bf16 sA[2][64][64];   // [buf][m][k]
    __shared__ __bf16 sB[2][64][64];   // [buf][n][k] (K contiguous per column)

    const int tid  = threadIdx.x;
    const int lane = tid & 31;
    const int wave = tid >> 5;
    const int wm   = wave & 3;          // M sub-tile 0..3
    const int wn   = wave >> 2;         // N sub-tile 0..1
    const int m0   = blockIdx.y * 64;
    const int n0   = blockIdx.x * 64;

    v8f acc0 = {0.f,0.f,0.f,0.f,0.f,0.f,0.f,0.f};
    v8f acc1 = {0.f,0.f,0.f,0.f,0.f,0.f,0.f,0.f};

    const int row = lane & 15;
    const int hi  = lane >> 4;          // lane half
    const int ka  = hi ? 16 : 0;        // A frag: K 0..15 vs 16..31
    const int kb  = hi ? 16 : 0;        // B frag: K 0..15 vs 16..31

    // per-thread staging coordinates (loop invariant)
    const int arr[4] = { (tid + 0)   >> 4, (tid + 256) >> 4,
                         (tid + 512) >> 4, (tid + 768) >> 4 };
    const int akg    = (tid & 15) << 2;

    float4 ra[4], rb[4];

    // phase 1: issue all global loads for tile at k0 into registers
    auto load_tile = [&](int k0) {
        #pragma unroll
        for (int g = 0; g < 4; ++g) {
            int gm = m0 + arr[g]; if (gm >= M) gm = M - 1;   // clamp; discarded at store
            ra[g] = *(const float4*)&A[(size_t)gm * K + k0 + akg];
        }
        if (transB) {
            #pragma unroll
            for (int g = 0; g < 4; ++g) {
                int gn = n0 + arr[g]; if (gn >= N) gn = N - 1;
                rb[g] = *(const float4*)&B[(size_t)gn * KB + k0 + akg];
            }
        } else if (N >= 4) {
            #pragma unroll
            for (int g = 0; g < 4; ++g) {
                int gk  = k0 + arr[g];
                int gkc = (gk < KB) ? gk : (KB - 1);
                rb[g] = *(const float4*)&B[(size_t)gkc * N + n0 + akg];
            }
        } else {
            #pragma unroll
            for (int g = 0; g < 4; ++g) {
                int gk  = k0 + arr[g];
                int gkc = (gk < KB) ? gk : (KB - 1);
                float4 v = {0.f, 0.f, 0.f, 0.f};
                if (gk < KB) {
                    if (n0 + akg + 0 < N) v.x = B[(size_t)gkc * N + n0 + akg + 0];
                    if (n0 + akg + 1 < N) v.y = B[(size_t)gkc * N + n0 + akg + 1];
                    if (n0 + akg + 2 < N) v.z = B[(size_t)gkc * N + n0 + akg + 2];
                    if (n0 + akg + 3 < N) v.w = B[(size_t)gkc * N + n0 + akg + 3];
                }
                rb[g] = v;
            }
        }
    };

    // phase 2: convert f32 -> bf16 and store into LDS buffer
    auto store_tile = [&](int k0, int buf) {
        #pragma unroll
        for (int g = 0; g < 4; ++g) {
            v4bf w;
            w[0] = (__bf16)ra[g].x; w[1] = (__bf16)ra[g].y;
            w[2] = (__bf16)ra[g].z; w[3] = (__bf16)ra[g].w;
            *(v4bf*)&sA[buf][arr[g]][akg] = w;
        }
        if (transB) {
            #pragma unroll
            for (int g = 0; g < 4; ++g) {
                v4bf w;
                w[0] = (__bf16)rb[g].x; w[1] = (__bf16)rb[g].y;
                w[2] = (__bf16)rb[g].z; w[3] = (__bf16)rb[g].w;
                *(v4bf*)&sB[buf][arr[g]][akg] = w;
            }
        } else {
            // scatter along n (4 columns), zero K-padding rows
            #pragma unroll
            for (int g = 0; g < 4; ++g) {
                int k  = arr[g];
                int gk = k0 + k;
                float z = (gk < KB) ? 1.f : 0.f;
                sB[buf][akg + 0][k] = (__bf16)(rb[g].x * z);
                sB[buf][akg + 1][k] = (__bf16)(rb[g].y * z);
                sB[buf][akg + 2][k] = (__bf16)(rb[g].z * z);
                sB[buf][akg + 3][k] = (__bf16)(rb[g].w * z);
            }
        }
    };

    // prologue: stage tile 0
    load_tile(0);
    store_tile(0, 0);
    __syncthreads();

    int buf = 0;
    for (int k0 = 0; k0 < K; k0 += 64) {
        const bool has_next = (k0 + 64 < K);
        if (has_next) load_tile(k0 + 64);          // loads in flight during compute

        // ---- 2 K-subchunks x 2 N-subtiles = 4 WMMAs per round ----
        #pragma unroll
        for (int ks = 0; ks < 64; ks += 32) {
            v16bf a;
            const __bf16* ap = &sA[buf][wm * 16 + row][ks];
            #pragma unroll
            for (int i = 0; i < 8; ++i) {
                a[i]     = ap[ka + i];
                a[8 + i] = ap[ka + 8 + i];
            }
            const __bf16* bp0 = &sB[buf][wn * 32 + row][ks + kb];
            const __bf16* bp1 = &sB[buf][wn * 32 + 16 + row][ks + kb];
            v16bf b0, b1;
            #pragma unroll
            for (int i = 0; i < 16; ++i) { b0[i] = bp0[i]; b1[i] = bp1[i]; }
            acc0 = __builtin_amdgcn_wmma_f32_16x16x32_bf16(
                       false, a, false, b0, (short)0, acc0, false, false);
            acc1 = __builtin_amdgcn_wmma_f32_16x16x32_bf16(
                       false, a, false, b1, (short)0, acc1, false, false);
        }

        if (has_next) {
            store_tile(k0 + 64, buf ^ 1);          // waits loads, fills other buffer
            __syncthreads();
            buf ^= 1;
        }
    }

    // ---- store: VGPR e -> M = e (+8 for upper lane half), N = lane&15 ----
    const int col = lane & 15;
    #pragma unroll
    for (int ns = 0; ns < 2; ++ns) {
        int gn = n0 + wn * 32 + ns * 16 + col;
        if (gn < N) {
            float bv = bias ? bias[gn] : 0.f;
            const v8f& acc = ns ? acc1 : acc0;
            #pragma unroll
            for (int e = 0; e < 8; ++e) {
                int gm = m0 + wm * 16 + hi * 8 + e;
                if (gm < M) {
                    float v = acc[e] + bv;
                    if (relu) v = fmaxf(v, 0.f);
                    C[(size_t)gm * N + gn] = v;
                }
            }
        }
    }
}

// ---------------------------------------------------------------------------
// gi = s[t] * Wih[:,0] + bih   (scalar-input GRU projection, in_size==1)
// ---------------------------------------------------------------------------
__global__ void gvae_gi_scalar(const float* __restrict__ s,
                               const float* __restrict__ Wih,
                               const float* __restrict__ bih,
                               float* __restrict__ gi, int T_, int H3)
{
    int i = blockIdx.x * 256 + threadIdx.x;
    if (i >= T_ * H3) return;
    int t = i / H3, j = i - t * H3;
    gi[i] = s[t] * Wih[j] + bih[j];
}

// ---------------------------------------------------------------------------
// Sequential GRU scan: one persistent workgroup, h in LDS, Whh streamed
// from L2 (all weights fit in the 192MB L2). 4096 serial steps.
// ---------------------------------------------------------------------------
__global__ __launch_bounds__(1024)
void gvae_gru_scan(const float* __restrict__ gi, const float* __restrict__ Whh,
                   const float* __restrict__ bhh, float* __restrict__ hs,
                   int T_, int H)
{
    __shared__ float h_cur[512];
    __shared__ float gh[1536];
    const int tid = threadIdx.x;
    const int H3  = 3 * H;
    const int H4  = H >> 2;

    for (int j = tid; j < H; j += 1024) h_cur[j] = 0.f;
    __syncthreads();

    for (int t = 0; t < T_; ++t) {
        // gh = h @ Whh.T + bhh  (float4 b128 loads from L2 and LDS)
        for (int j = tid; j < H3; j += 1024) {
            const float4* w = (const float4*)(Whh + (size_t)j * H);
            float acc = bhh[j];
            for (int k = 0; k < H4; ++k) {
                float4 wv = w[k];
                float4 hv = *(const float4*)&h_cur[k << 2];
                acc += wv.x * hv.x + wv.y * hv.y + wv.z * hv.z + wv.w * hv.w;
            }
            gh[j] = acc;
        }
        __syncthreads();
        const float* git = gi + (size_t)t * H3;
        for (int j = tid; j < H; j += 1024) {
            float r  = 1.f / (1.f + __expf(-(git[j]         + gh[j])));
            float z  = 1.f / (1.f + __expf(-(git[H + j]     + gh[H + j])));
            float n  = tanhf(git[2 * H + j] + r * gh[2 * H + j]);
            float hn = (1.f - z) * n + z * h_cur[j];
            h_cur[j] = hn;
            hs[(size_t)t * H + j] = hn;
        }
        __syncthreads();
    }
}

// ---------------------------------------------------------------------------
// Elementwise kernels
// ---------------------------------------------------------------------------
__device__ __forceinline__ float gvae_softplus(float x) {
    return fmaxf(x, 0.f) + log1pf(__expf(-fabsf(x)));
}

__global__ void gvae_zs(const float* __restrict__ mv, const float* __restrict__ eps,
                        float* __restrict__ pm, float* __restrict__ ps,
                        float* __restrict__ zs, int n)
{
    int i = blockIdx.x * 256 + threadIdx.x;
    if (i >= n) return;
    int t = i >> 7, j = i & 127;                 // Z_SIZE = 128
    float m   = mv[t * 256 + j];
    float raw = mv[t * 256 + 128 + j];
    float sp  = gvae_softplus(raw);
    pm[i] = m;
    ps[i] = sp;
    zs[i] = m + sp * eps[i];
}

__global__ void gvae_prior_split(const float* __restrict__ pmv,
                                 float* __restrict__ pm, float* __restrict__ ps, int n)
{
    int i = blockIdx.x * 256 + threadIdx.x;
    if (i >= n) return;
    int t = i >> 7, j = i & 127;
    pm[i] = pmv[t * 256 + j];
    ps[i] = gvae_softplus(pmv[t * 256 + 128 + j]);
}

// dst[t*ldd + c0 + j] = src[(t+r0)*lds + j]   for t<rows, j<cols
__global__ void gvae_copy_cols(float* __restrict__ dst, int ldd, int c0,
                               const float* __restrict__ src, int lds_, int r0,
                               int rows, int cols)
{
    int i = blockIdx.x * 256 + threadIdx.x;
    if (i >= rows * cols) return;
    int t = i / cols, j = i - t * cols;
    dst[(size_t)t * ldd + c0 + j] = src[(size_t)(t + r0) * lds_ + j];
}

__global__ void gvae_zero(float* __restrict__ dst, int n)
{
    int i = blockIdx.x * 256 + threadIdx.x;
    if (i < n) dst[i] = 0.f;
}

// ---------------------------------------------------------------------------
// Host-side orchestration
// ---------------------------------------------------------------------------
static inline void launch_gemm(const float* A, const float* B, const float* bias,
                               float* C, int M, int N, int K, int KB,
                               int transB, int relu, hipStream_t s)
{
    dim3 grid((N + 63) / 64, (M + 63) / 64);
    gvae_wmma_gemm<<<grid, 256, 0, s>>>(A, B, bias, C, M, N, K, KB, transB, relu);
}

static inline dim3 g1(int n) { return dim3((n + 255) / 256); }

extern "C" void kernel_launch(void* const* d_in, const int* in_sizes, int n_in,
                              void* d_out, int out_size, void* d_ws, size_t ws_size,
                              hipStream_t stream)
{
    (void)in_sizes; (void)n_in; (void)out_size; (void)ws_size;

    const float* y   = (const float*)d_in[0];
    const float* p   = (const float*)d_in[1];
    const float* x   = (const float*)d_in[2];
    const float* eps = (const float*)d_in[3];

    // gru params: Wih, Whh, bih, bhh
    const float* Wih_y  = (const float*)d_in[4];
    const float* Whh_y  = (const float*)d_in[5];
    const float* bih_y  = (const float*)d_in[6];
    const float* bhh_y  = (const float*)d_in[7];
    const float* Wih_x  = (const float*)d_in[8];
    const float* Whh_x  = (const float*)d_in[9];
    const float* bih_x  = (const float*)d_in[10];
    const float* bhh_x  = (const float*)d_in[11];
    const float* Wih_p  = (const float*)d_in[12];
    const float* Whh_p  = (const float*)d_in[13];
    const float* bih_p  = (const float*)d_in[14];
    const float* bhh_p  = (const float*)d_in[15];
    const float* Wih_q  = (const float*)d_in[16];   // gru_z_post
    const float* Whh_q  = (const float*)d_in[17];
    const float* bih_q  = (const float*)d_in[18];
    const float* bhh_q  = (const float*)d_in[19];
    const float* Wih_z  = (const float*)d_in[20];   // gru_z_prior
    const float* Whh_z  = (const float*)d_in[21];
    const float* bih_z  = (const float*)d_in[22];
    const float* bhh_z  = (const float*)d_in[23];
    const float* Wzp    = (const float*)d_in[24];   // z_post.W (512,256)
    const float* bzp    = (const float*)d_in[25];
    const float* zpr_W0 = (const float*)d_in[26];   // z_prior mlp
    const float* zpr_b0 = (const float*)d_in[27];
    const float* zpr_Wo = (const float*)d_in[28];
    const float* zpr_bo = (const float*)d_in[29];
    const float* ye_W0  = (const float*)d_in[30];   // y_emission
    const float* ye_b0  = (const float*)d_in[31];
    const float* ye_Wo  = (const float*)d_in[32];
    const float* ye_bo  = (const float*)d_in[33];
    const float* y2_W0  = (const float*)d_in[34];   // y_emission2 (257,2048)
    const float* y2_b0  = (const float*)d_in[35];
    const float* y2_Wo  = (const float*)d_in[36];
    const float* y2_bo  = (const float*)d_in[37];
    const float* xe_W0  = (const float*)d_in[38];   // x_emission
    const float* xe_b0  = (const float*)d_in[39];
    const float* xe_Wo  = (const float*)d_in[40];
    const float* xe_bo  = (const float*)d_in[41];
    const float* pe_W0  = (const float*)d_in[42];   // p_emission (1024,2048)
    const float* pe_b0  = (const float*)d_in[43];
    const float* pe_Wo  = (const float*)d_in[44];
    const float* pe_bo  = (const float*)d_in[45];

    // ---- workspace carve-up (floats) ----
    float* ws = (float*)d_ws;
    size_t o = 0;
    float* gi_y  = ws + o; o += (size_t)T_ALL * 768;
    float* gi_x  = ws + o; o += (size_t)T_ALL * 768;
    float* gi_p  = ws + o; o += (size_t)T_ALL * 1536;
    float* gi_q  = ws + o; o += (size_t)T_ALL * 1536;
    float* gt    = ws + o; o += (size_t)T_ALL * 512;
    float* mv    = ws + o; o += (size_t)T_M1 * 256;
    float* gi_z  = ws + o; o += (size_t)T_M1 * 1536;
    float* hprev = ws + o; o += (size_t)T_M1 * 512;
    float* Hbuf  = ws + o; o += (size_t)T_M1 * 2048;
    float* cat   = ws + o; o += (size_t)T_M1 * 1024;
    float* pmv   = ws + o; o += (size_t)T_M1 * 256;

    // ---- output carve-up (return-tuple order) ----
    float* out  = (float*)d_out;
    float* o_zs = out;
    float* o_pm = o_zs + (size_t)T_M1 * 128;
    float* o_ps = o_pm + (size_t)T_M1 * 128;
    float* o_qm = o_ps + (size_t)T_M1 * 128;
    float* o_qs = o_qm + (size_t)T_M1 * 128;
    float* o_hz = o_qs + (size_t)T_M1 * 128;
    float* o_hx = o_hz + (size_t)T_M1 * 512;
    float* o_hy = o_hx + (size_t)T_ALL * 256;
    float* o_hp = o_hy + (size_t)T_ALL * 256;
    float* o_xh = o_hp + (size_t)T_ALL * 512;
    float* o_yh = o_xh + (size_t)T_M1;
    float* o_ph = o_yh + (size_t)T_M1;
    float* o_yn = o_ph + (size_t)T_M1 * 64;

    // 1. input projections
    gvae_gi_scalar<<<g1(T_ALL * 768), 256, 0, stream>>>(y, Wih_y, bih_y, gi_y, T_ALL, 768);
    gvae_gi_scalar<<<g1(T_ALL * 768), 256, 0, stream>>>(x, Wih_x, bih_x, gi_x, T_ALL, 768);
    launch_gemm(p, Wih_p, bih_p, gi_p, T_ALL, 1536, 64, 64, 1, 0, stream);
    launch_gemm(p, Wih_q, bih_q, gi_q, T_ALL, 1536, 64, 64, 1, 0, stream);

    // 2. independent sequential scans
    gvae_gru_scan<<<1, 1024, 0, stream>>>(gi_y, Whh_y, bhh_y, o_hy, T_ALL, 256);
    gvae_gru_scan<<<1, 1024, 0, stream>>>(gi_x, Whh_x, bhh_x, o_hx, T_ALL, 256);
    gvae_gru_scan<<<1, 1024, 0, stream>>>(gi_p, Whh_p, bhh_p, o_hp, T_ALL, 512);
    gvae_gru_scan<<<1, 1024, 0, stream>>>(gi_q, Whh_q, bhh_q, gt,   T_ALL, 512);

    // 3. mv = gt[1:] @ z_post.W + b
    launch_gemm(gt + 512, Wzp, bzp, mv, T_M1, 256, 512, 512, 0, 0, stream);

    // 4. posterior split + reparameterized sample
    gvae_zs<<<g1(T_M1 * 128), 256, 0, stream>>>(mv, eps, o_qm, o_qs, o_zs, T_M1 * 128);

    // 5. gi for gru_z_prior, then its scan -> hz
    launch_gemm(o_zs, Wih_z, bih_z, gi_z, T_M1, 1536, 128, 128, 1, 0, stream);
    gvae_gru_scan<<<1, 1024, 0, stream>>>(gi_z, Whh_z, bhh_z, o_hz, T_M1, 512);

    // 6. h_prev = [0; hz[:-1]]
    gvae_zero<<<g1(512), 256, 0, stream>>>(hprev, 512);
    gvae_copy_cols<<<g1((T_M1 - 1) * 512), 256, 0, stream>>>(hprev + 512, 512, 0, o_hz, 512, 0, T_M1 - 1, 512);

    // 7. z_prior MLP -> prior mean/std
    launch_gemm(hprev, zpr_W0, zpr_b0, Hbuf, T_M1, 2048, 512, 512, 0, 1, stream);
    launch_gemm(Hbuf, zpr_Wo, zpr_bo, pmv, T_M1, 256, 2048, 2048, 0, 0, stream);
    gvae_prior_split<<<g1(T_M1 * 128), 256, 0, stream>>>(pmv, o_pm, o_ps, T_M1 * 128);

    // 8. p_emission: [hz, hp[1:]] (K=1024) -> 2048 -> 64
    gvae_copy_cols<<<g1(T_M1 * 512), 256, 0, stream>>>(cat, 1024, 0,   o_hz, 512, 0, T_M1, 512);
    gvae_copy_cols<<<g1(T_M1 * 512), 256, 0, stream>>>(cat, 1024, 512, o_hp, 512, 1, T_M1, 512);
    launch_gemm(cat, pe_W0, pe_b0, Hbuf, T_M1, 2048, 1024, 1024, 0, 1, stream);
    launch_gemm(Hbuf, pe_Wo, pe_bo, o_ph, T_M1, 64, 2048, 2048, 0, 0, stream);

    // 9. x_emission: [hz, hx[1:]] (K=768) -> 2048 -> 1
    gvae_copy_cols<<<g1(T_M1 * 512), 256, 0, stream>>>(cat, 768, 0,   o_hz, 512, 0, T_M1, 512);
    gvae_copy_cols<<<g1(T_M1 * 256), 256, 0, stream>>>(cat, 768, 512, o_hx, 256, 1, T_M1, 256);
    launch_gemm(cat, xe_W0, xe_b0, Hbuf, T_M1, 2048, 768, 768, 0, 1, stream);
    launch_gemm(Hbuf, xe_Wo, xe_bo, o_xh, T_M1, 1, 2048, 2048, 0, 0, stream);

    // 10. y_emission: [hz, hy[1:]] (K=768) -> 2048 -> 1  (= y_nox)
    gvae_copy_cols<<<g1(T_M1 * 512), 256, 0, stream>>>(cat, 768, 0,   o_hz, 512, 0, T_M1, 512);
    gvae_copy_cols<<<g1(T_M1 * 256), 256, 0, stream>>>(cat, 768, 512, o_hy, 256, 1, T_M1, 256);
    launch_gemm(cat, ye_W0, ye_b0, Hbuf, T_M1, 2048, 768, 768, 0, 1, stream);
    launch_gemm(Hbuf, ye_Wo, ye_bo, o_yn, T_M1, 1, 2048, 2048, 0, 0, stream);

    // 11. y_emission2: [y_nox, hx[1:]] (K=257, padded to 320) -> 2048 -> 1
    gvae_zero<<<g1(T_M1 * 320), 256, 0, stream>>>(cat, T_M1 * 320);
    gvae_copy_cols<<<g1(T_M1), 256, 0, stream>>>(cat, 320, 0, o_yn, 1, 0, T_M1, 1);
    gvae_copy_cols<<<g1(T_M1 * 256), 256, 0, stream>>>(cat, 320, 1, o_hx, 256, 1, T_M1, 256);
    launch_gemm(cat, y2_W0, y2_b0, Hbuf, T_M1, 2048, 320, 257, 0, 1, stream);
    launch_gemm(Hbuf, y2_Wo, y2_bo, o_yh, T_M1, 1, 2048, 2048, 0, 0, stream);
}
